// MultiHeadAttention_62792421867720
// MI455X (gfx1250) — matrix operand order repaired and checked by reading, not compiled, not run on hardware
//
#include <hip/hip_runtime.h>
#include <hip/hip_bf16.h>

#define DMODEL 1024
#define NHEADS 16
#define DK     64
#define BATCH  2
#define SEQ    2048

typedef _Float16 h16;
typedef __attribute__((ext_vector_type(16))) _Float16 v16h;
typedef __attribute__((ext_vector_type(8)))  _Float16 v8h;
typedef __attribute__((ext_vector_type(8)))  float    v8f;

#define WMMA_F16(a, b, c) \
  __builtin_amdgcn_wmma_f32_16x16x32_f16(false, (a), false, (b), (short)0, (c), false, false)

// ---------------------------------------------------------------------------
// A fragment (M=16 x K=32, f16) from a row-major matrix, leading dim ld.
// Layout: lane L holds row m = L&15; lanes 0-15 cover K=0-15,
// lanes 16-31 cover K=16-31 (kbase = (L>>4)*16).
// ---------------------------------------------------------------------------
__device__ __forceinline__ v16h frag_a_rowmajor(const h16* p, int ld) {
  const int lane = threadIdx.x & 31;
  const int m  = lane & 15;
  const int kb = (lane >> 4) << 4;          // 0 or 16
  const h16* r = p + (size_t)m * ld + kb;
  v8h lo = *(const v8h*)(r);                // K = kb .. kb+7
  v8h hi = *(const v8h*)(r + 8);            // K = kb+8 .. kb+15
  v16h a;
#pragma unroll
  for (int i = 0; i < 8; ++i) { a[i] = lo[i]; a[i + 8] = hi[i]; }
  return a;
}

// ---------------------------------------------------------------------------
// B fragment (K=32 x N=16, f16) where element (k,n) lives at p[n*ld + k]
// (natural for W[o][d], K[s][d], V^T[dk][s]).  Lane L holds column n = L&15;
// lanes 0-15 take K=0-15, lanes 16-31 take K=16-31 -> one 32B load per lane.
// ---------------------------------------------------------------------------
__device__ __forceinline__ v16h frag_b_nmajor(const h16* p, int ld) {
  const int lane = threadIdx.x & 31;
  const int n  = lane & 15;
  const int kb = (lane >> 4) << 4;          // 0 or 16
  return *(const v16h*)(p + (size_t)n * ld + kb);
}

// ---------------------------------------------------------------------------
// fp32 -> f16 conversion (grid-stride)
// ---------------------------------------------------------------------------
__global__ void cvt_f32_f16(const float* __restrict__ src, h16* __restrict__ dst, int n) {
  int i = blockIdx.x * blockDim.x + threadIdx.x;
  const int stride = gridDim.x * blockDim.x;
  for (; i < n; i += stride) dst[i] = (h16)src[i];
}

// ---------------------------------------------------------------------------
// QKV projection + RoPE.  2x2 register-blocked: one 32x32 output tile per
// wave (4 fragment loads feed 4 wmmas per K step).
//   q[s,o] = sum_d X[s,d] * W[o,d]
// which = 0:Q (RoPE), 1:K (RoPE), 2:V (stored transposed [b,h,dk,s])
// ---------------------------------------------------------------------------
__global__ void qkv_rope_kernel(const h16* __restrict__ X,
                                const h16* __restrict__ Wq,
                                const h16* __restrict__ Wk,
                                const h16* __restrict__ Wv,
                                const long long* __restrict__ pos,
                                h16* __restrict__ Q, h16* __restrict__ K,
                                h16* __restrict__ VT) {
  const int lane = threadIdx.x & 31;
  const int wave = threadIdx.x >> 5;
  const int s0 = blockIdx.x * 32;                    // 32 rows per wave
  const int o0 = (blockIdx.y * 4 + wave) * 32;       // 32 cols per wave
  const int b     = blockIdx.z / 3;
  const int which = blockIdx.z % 3;
  const h16* W = (which == 0) ? Wq : ((which == 1) ? Wk : Wv);
  const h16* Ar = X + (size_t)(b * SEQ + s0) * DMODEL;
  const h16* Br = W + (size_t)o0 * DMODEL;

  v8f acc[2][2] = {};
  for (int k = 0; k < DMODEL; k += 32) {
    __builtin_prefetch(Ar + k + 64, 0, 1);           // global_prefetch_b8
    __builtin_prefetch(Br + k + 64, 0, 1);
    v16h a0 = frag_a_rowmajor(Ar + k, DMODEL);
    v16h a1 = frag_a_rowmajor(Ar + (size_t)16 * DMODEL + k, DMODEL);
    v16h b0 = frag_b_nmajor(Br + k, DMODEL);
    v16h b1 = frag_b_nmajor(Br + (size_t)16 * DMODEL + k, DMODEL);
    acc[0][0] = WMMA_F16(a0, b0, acc[0][0]);
    acc[0][1] = WMMA_F16(a0, b1, acc[0][1]);
    acc[1][0] = WMMA_F16(a1, b0, acc[1][0]);
    acc[1][1] = WMMA_F16(a1, b1, acc[1][1]);
  }

  const int n   = lane & 15;
  const int mhi = (lane >> 4) << 3;                  // row offset 0 or 8

#pragma unroll
  for (int ct = 0; ct < 2; ++ct) {
    const int o  = o0 + ct * 16 + n;
    const int h  = o >> 6;
    const int dk = o & 63;
    if (which == 2) {
      // V stored transposed: VT[b][h][dk][s]
#pragma unroll
      for (int rt = 0; rt < 2; ++rt) {
#pragma unroll
        for (int r = 0; r < 8; ++r) {
          const int s = s0 + rt * 16 + r + mhi;
          VT[((size_t)(b * NHEADS + h) * DK + dk) * SEQ + s] = (h16)acc[rt][ct][r];
        }
      }
    } else {
      h16* dst = (which == 0) ? Q : K;
      // inv_freq = theta^(-2i/DK), i = dk>>1 ; column parity == lane parity
      const float invf = expf((float)(2 * (dk >> 1)) * (-logf(10000.0f) / (float)DK));
      const float sgn  = (dk & 1) ? 1.0f : -1.0f;
#pragma unroll
      for (int rt = 0; rt < 2; ++rt) {
#pragma unroll
        for (int r = 0; r < 8; ++r) {
          const int s = s0 + rt * 16 + r + mhi;
          const float ang = (float)pos[s] * invf;
          const float partner = __shfl_xor(acc[rt][ct][r], 1, 32);  // pair column
          const float v = acc[rt][ct][r] * cosf(ang) + sgn * partner * sinf(ang);
          dst[((size_t)(b * NHEADS + h) * SEQ + s) * DK + dk] = (h16)v;
        }
      }
    }
  }
}

// ---------------------------------------------------------------------------
// Flash attention: one wave per (b, h, 16-row q tile); 32 keys per step.
// S = Q K^T via 4 wmma (K=64), online softmax in registers (cross-lane
// reductions within 16-lane halves), P staged through per-wave LDS for the
// C-layout -> A-layout transpose, then O += P V via 4 wmma (N = 4x16 dims).
// ---------------------------------------------------------------------------
__global__ void attn_kernel(const h16* __restrict__ Q, const h16* __restrict__ Km,
                            const h16* __restrict__ VT, h16* __restrict__ O) {
  __shared__ __align__(32) h16 Plds[4][16][32];
  const int lane = threadIdx.x & 31;
  const int wave = threadIdx.x >> 5;
  const int bh = blockIdx.y;
  const int b = bh / NHEADS, h = bh % NHEADS;
  const int q0 = (blockIdx.x * 4 + wave) * 16;

  const h16* Qb = Q  + ((size_t)bh * SEQ + q0) * DK;
  const h16* Kb = Km + (size_t)bh * SEQ * DK;
  const h16* Vb = VT + (size_t)bh * DK * SEQ;

  const v16h qa0 = frag_a_rowmajor(Qb, DK);        // d = 0..31
  const v16h qa1 = frag_a_rowmajor(Qb + 32, DK);   // d = 32..63

  const int n   = lane & 15;
  const int mhi = (lane >> 4) << 3;

  float mrun[8], lrun[8];
  v8f o0t = {}, o1t = {}, o2t = {}, o3t = {};
#pragma unroll
  for (int r = 0; r < 8; ++r) { mrun[r] = -1e30f; lrun[r] = 0.0f; }

  for (int kb = 0; kb <= q0 + 15; kb += 32) {
    __builtin_prefetch(Kb + (size_t)(kb + 32) * DK, 0, 1);
    __builtin_prefetch(Vb + kb + 32, 0, 1);
    // ---- scores: two 16x16 tiles covering 32 keys, K-dim = 64 ----
    v8f s0 = {}, s1 = {};
    s0 = WMMA_F16(qa0, frag_b_nmajor(Kb + (size_t)kb * DK,             DK), s0);
    s0 = WMMA_F16(qa1, frag_b_nmajor(Kb + (size_t)kb * DK + 32,        DK), s0);
    s1 = WMMA_F16(qa0, frag_b_nmajor(Kb + (size_t)(kb + 16) * DK,      DK), s1);
    s1 = WMMA_F16(qa1, frag_b_nmajor(Kb + (size_t)(kb + 16) * DK + 32, DK), s1);

    // ---- scale + causal mask + online softmax ----
#pragma unroll
    for (int r = 0; r < 8; ++r) {
      const int row = q0 + r + mhi;
      float a0 = s0[r] * 0.125f;                 // 1/sqrt(64)
      float a1 = s1[r] * 0.125f;
      if (kb + n      > row) a0 = -1e30f;
      if (kb + 16 + n > row) a1 = -1e30f;
      float rm = fmaxf(a0, a1);
#pragma unroll
      for (int off = 1; off < 16; off <<= 1) rm = fmaxf(rm, __shfl_xor(rm, off, 32));
      const float mn = fmaxf(mrun[r], rm);
      const float alpha = __expf(mrun[r] - mn);
      mrun[r] = mn;
      const float p0 = __expf(a0 - mn);
      const float p1 = __expf(a1 - mn);
      float rs = p0 + p1;
#pragma unroll
      for (int off = 1; off < 16; off <<= 1) rs += __shfl_xor(rs, off, 32);
      lrun[r] = lrun[r] * alpha + rs;
      o0t[r] *= alpha; o1t[r] *= alpha; o2t[r] *= alpha; o3t[r] *= alpha;
      Plds[wave][r + mhi][n]      = (h16)p0;
      Plds[wave][r + mhi][16 + n] = (h16)p1;
    }
    asm volatile("s_wait_dscnt 0x0" ::: "memory");   // LDS writes visible to lane reads

    // ---- P as A-fragment (16x32) from LDS ----
    v16h pa;
    {
      const int m   = lane & 15;
      const int kb8 = (lane >> 4) << 4;
#pragma unroll
      for (int j = 0; j < 8; ++j) {
        pa[j]     = Plds[wave][m][kb8 + j];
        pa[j + 8] = Plds[wave][m][kb8 + 8 + j];
      }
    }
    // ---- O += P * V  (V^T stored [dk][s] -> contiguous B fragments) ----
    o0t = WMMA_F16(pa, frag_b_nmajor(Vb + (size_t) 0 * SEQ + kb, SEQ), o0t);
    o1t = WMMA_F16(pa, frag_b_nmajor(Vb + (size_t)16 * SEQ + kb, SEQ), o1t);
    o2t = WMMA_F16(pa, frag_b_nmajor(Vb + (size_t)32 * SEQ + kb, SEQ), o2t);
    o3t = WMMA_F16(pa, frag_b_nmajor(Vb + (size_t)48 * SEQ + kb, SEQ), o3t);
  }

  // ---- normalize and store attention output as [b][s][h*64+dk] f16 ----
#pragma unroll
  for (int r = 0; r < 8; ++r) {
    const int s = q0 + r + mhi;
    const float inv = 1.0f / lrun[r];
    const size_t base = ((size_t)b * SEQ + s) * DMODEL + (size_t)h * DK;
    O[base +  0 + n] = (h16)(o0t[r] * inv);
    O[base + 16 + n] = (h16)(o1t[r] * inv);
    O[base + 32 + n] = (h16)(o2t[r] * inv);
    O[base + 48 + n] = (h16)(o3t[r] * inv);
  }
}

// ---------------------------------------------------------------------------
// Output projection: out[s,o] = sum_d A[s,d] * Wo[o,d]  (f32 stores)
// 2x2 register-blocked like the QKV GEMM.
// ---------------------------------------------------------------------------
__global__ void proj_out_kernel(const h16* __restrict__ A, const h16* __restrict__ Wo,
                                float* __restrict__ out) {
  const int lane = threadIdx.x & 31;
  const int wave = threadIdx.x >> 5;
  const int m0 = blockIdx.x * 32;
  const int o0 = (blockIdx.y * 4 + wave) * 32;
  const h16* Ar = A  + (size_t)m0 * DMODEL;
  const h16* Br = Wo + (size_t)o0 * DMODEL;

  v8f acc[2][2] = {};
  for (int k = 0; k < DMODEL; k += 32) {
    __builtin_prefetch(Ar + k + 64, 0, 1);
    __builtin_prefetch(Br + k + 64, 0, 1);
    v16h a0 = frag_a_rowmajor(Ar + k, DMODEL);
    v16h a1 = frag_a_rowmajor(Ar + (size_t)16 * DMODEL + k, DMODEL);
    v16h b0 = frag_b_nmajor(Br + k, DMODEL);
    v16h b1 = frag_b_nmajor(Br + (size_t)16 * DMODEL + k, DMODEL);
    acc[0][0] = WMMA_F16(a0, b0, acc[0][0]);
    acc[0][1] = WMMA_F16(a0, b1, acc[0][1]);
    acc[1][0] = WMMA_F16(a1, b0, acc[1][0]);
    acc[1][1] = WMMA_F16(a1, b1, acc[1][1]);
  }
  const int n   = lane & 15;
  const int mhi = (lane >> 4) << 3;
#pragma unroll
  for (int rt = 0; rt < 2; ++rt)
#pragma unroll
    for (int ct = 0; ct < 2; ++ct)
#pragma unroll
      for (int r = 0; r < 8; ++r)
        out[(size_t)(m0 + rt * 16 + r + mhi) * DMODEL + o0 + ct * 16 + n] = acc[rt][ct][r];
}

// ---------------------------------------------------------------------------
// Launch: 48 MB workspace layout (in f16 elements):
//   X16 | Wq16 Wk16 Wv16 Wo16 | Q16 | K16 | VT16 | O16
// ---------------------------------------------------------------------------
extern "C" void kernel_launch(void* const* d_in, const int* in_sizes, int n_in,
                              void* d_out, int out_size, void* d_ws, size_t ws_size,
                              hipStream_t stream) {
  (void)in_sizes; (void)n_in; (void)out_size; (void)ws_size;
  const float*      x   = (const float*)d_in[0];
  const long long*  pos = (const long long*)d_in[1];
  const float*      wq  = (const float*)d_in[2];
  const float*      wk  = (const float*)d_in[3];
  const float*      wv  = (const float*)d_in[4];
  const float*      wo  = (const float*)d_in[5];
  float* out = (float*)d_out;

  h16* ws = (h16*)d_ws;
  const size_t NX = (size_t)BATCH * SEQ * DMODEL;   // 4,194,304
  const size_t NW = (size_t)DMODEL * DMODEL;        // 1,048,576
  h16* X16  = ws;
  h16* Wq16 = X16  + NX;
  h16* Wk16 = Wq16 + NW;
  h16* Wv16 = Wk16 + NW;
  h16* Wo16 = Wv16 + NW;
  h16* Q16  = Wo16 + NW;
  h16* K16  = Q16  + NX;
  h16* VT16 = K16  + NX;
  h16* O16  = VT16 + NX;

  cvt_f32_f16<<<dim3(4096), dim3(256), 0, stream>>>(x,  X16,  (int)NX);
  cvt_f32_f16<<<dim3(1024), dim3(256), 0, stream>>>(wq, Wq16, (int)NW);
  cvt_f32_f16<<<dim3(1024), dim3(256), 0, stream>>>(wk, Wk16, (int)NW);
  cvt_f32_f16<<<dim3(1024), dim3(256), 0, stream>>>(wv, Wv16, (int)NW);
  cvt_f32_f16<<<dim3(1024), dim3(256), 0, stream>>>(wo, Wo16, (int)NW);

  // (S/32, D/(32*4 waves), B*3 matrices), 4 waves per block
  qkv_rope_kernel<<<dim3(SEQ / 32, DMODEL / 128, BATCH * 3), dim3(128), 0, stream>>>(
      X16, Wq16, Wk16, Wv16, pos, Q16, K16, VT16);

  // (S/(16*4 waves), B*H), 4 waves per block
  attn_kernel<<<dim3(SEQ / 64, BATCH * NHEADS), dim3(128), 0, stream>>>(
      Q16, K16, VT16, O16);

  // (B*S/32, D/(32*4 waves))
  proj_out_kernel<<<dim3(BATCH * SEQ / 32, DMODEL / 128), dim3(128), 0, stream>>>(
      O16, Wo16, out);
}
